// GRUaroundLinearModel_34548716929351
// MI455X (gfx1250) — matrix operand order repaired and checked by reading, not compiled, not run on hardware
//
#include <hip/hip_runtime.h>
#include <hip/hip_bf16.h>
#include <math.h>

// Problem constants (from reference)
#define T_STEPS 65536
#define IN_DIM  128
#define H_DIM   256
#define G3      768   // 3*H
#define LIN_DIM 128

typedef __attribute__((ext_vector_type(16))) __bf16 v16bf;
typedef __attribute__((ext_vector_type(8)))  float  v8f;
typedef __attribute__((ext_vector_type(4)))  unsigned int v4u;
typedef __attribute__((ext_vector_type(4)))  int  v4i;
typedef __attribute__((ext_vector_type(8)))  int  v8i;

static __device__ __forceinline__ float sigmoidf_(float x) {
    return 1.0f / (1.0f + __expf(-x));
}

// ---------------------------------------------------------------------------
// TDM: DMA one row of 768 f32 (3KB) from global memory into LDS.
// D# group 0: count=1 | lds_addr | global_addr(57b) | type=2
// D# group 1: data_size=4B, tensor_dim0=tile_dim0=768, tensor_dim1=tile_dim1=1,
//             tensor_dim0_stride=768. Groups 2/3 zero (<=2D tensor).
// ---------------------------------------------------------------------------
static __device__ __forceinline__ void tdm_load_row(unsigned lds_addr,
                                                    const float* gptr) {
    const unsigned long long ga = (unsigned long long)(const void*)gptr;
    v4u g0;
    g0[0] = 1u;                                            // count=1 (valid)
    g0[1] = lds_addr;                                      // LDS byte address
    g0[2] = (unsigned)(ga & 0xffffffffull);                // global_addr[31:0]
    g0[3] = (unsigned)((ga >> 32) & 0x01ffffffull)         // global_addr[56:32]
          | (2u << 30);                                    // type=2 ("image")
    v8i g1;
    g1[0] = 0x00020000;        // workgroup_mask=0, data_size=2 (4 bytes)
    g1[1] = (int)0x03000000;   // tensor_dim0 = 768  (bits 63:48 here)
    g1[2] = 0x00010000;        // tensor_dim0 hi = 0, tensor_dim1 = 1 (lo16)
    g1[3] = (int)0x03000000;   // tensor_dim1 hi = 0, tile_dim0 = 768
    g1[4] = 0x00000001;        // tile_dim1 = 1, tile_dim2 = 0
    g1[5] = 768;               // tensor_dim0_stride lo32
    g1[6] = 0;                 // stride0 hi16 | stride1 lo16
    g1[7] = 0;                 // stride1 hi32
    const v4i z4 = {0, 0, 0, 0};
#if defined(__clang_major__) && (__clang_major__ >= 23)
    const v8i z8 = {0, 0, 0, 0, 0, 0, 0, 0};
    __builtin_amdgcn_tensor_load_to_lds(g0, g1, z4, z4, z8, 0);
#else
    __builtin_amdgcn_tensor_load_to_lds(g0, g1, z4, z4, 0);
#endif
}

// ---------------------------------------------------------------------------
// Kernel A: iu_all[T, 3H] = input_gru[T, IN] @ W_ih[3H, IN]^T + b[3H]
// One wave computes one 16x16 output tile via 4 chained bf16 WMMAs (K=128).
// ---------------------------------------------------------------------------
__global__ void __launch_bounds__(256)
gemm_iu_wmma(const float* __restrict__ x,     // [T, IN]
             const float* __restrict__ W,     // [3H, IN]
             const float* __restrict__ bias,  // [3H]
             float* __restrict__ iu)          // [T, 3H]
{
    const int lane = threadIdx.x & 31;
    const int wave = threadIdx.x >> 5;
    const int ttile = blockIdx.x * 8 + wave;   // 0..4095
    const int jtile = blockIdx.y;              // 0..47
    const int t0 = ttile * 16;
    const int j0 = jtile * 16;
    const int half = lane >> 4;                // lane group 0/1
    const int lid  = lane & 15;

    v8f c = {};

    // A: lane holds row M=lid, K = kb*32 + half*8 + {0..7} and +16+{0..7}
    const float* arow = x + (size_t)(t0 + lid) * IN_DIM + half * 8;
    // B: lane holds output column N=lid (row of W), K = kb*32 + half*8 + {0..7} and +16+{0..7}
    const float* brow = W + (size_t)(j0 + lid) * IN_DIM + half * 8;

#pragma unroll
    for (int kb = 0; kb < 4; ++kb) {
        const float* ap = arow + kb * 32;
        const float* bp = brow + kb * 32;
        v16bf a, b;
#pragma unroll
        for (int e = 0; e < 8; ++e) {
            a[e]     = (__bf16)ap[e];
            a[8 + e] = (__bf16)ap[16 + e];
        }
#pragma unroll
        for (int e = 0; e < 16; ++e) {
            b[e] = (__bf16)bp[e < 8 ? e : 8 + e];
        }
        c = __builtin_amdgcn_wmma_f32_16x16x32_bf16(
                false, a, false, b, (short)0, c, false, false);
    }

    const float bj = bias[j0 + lid];
#pragma unroll
    for (int r = 0; r < 8; ++r) {
        const int t = t0 + r + half * 8;       // M = r + 8*(lane/16)
        iu[(size_t)t * G3 + j0 + lid] = c[r] + bj;
    }
}

// ---------------------------------------------------------------------------
// Kernel B: pred[t] = dot(input_linear[t,:], W_lin) + b_lin
// ---------------------------------------------------------------------------
__global__ void __launch_bounds__(256)
pred_kernel(const float* __restrict__ xl, const float* __restrict__ Wl,
            const float* __restrict__ bl, float* __restrict__ pred)
{
    const int lane = threadIdx.x & 31;
    const int wave = threadIdx.x >> 5;
    const int t = blockIdx.x * 8 + wave;
    const float4 xv = ((const float4*)(xl + (size_t)t * LIN_DIM))[lane];
    const float4 wv = ((const float4*)Wl)[lane];
    float s = xv.x * wv.x + xv.y * wv.y + xv.z * wv.z + xv.w * wv.w;
#pragma unroll
    for (int off = 16; off > 0; off >>= 1)
        s += __shfl_xor(s, off, 32);
    if (lane == 0) pred[t] = s + bl[0];
}

// ---------------------------------------------------------------------------
// Kernel P: pre-pack W_hh[3H, H] into the bf16 WMMA A-operand lane layout:
//   Wb[((tile*8 + kb)*32 + lane)*16 + e] = bf16(W_hh[tile*16 + lane%16]
//                                                [kb*32 + (lane/16)*16 + e])
// ---------------------------------------------------------------------------
__global__ void __launch_bounds__(256)
pack_whh_wmma(const float* __restrict__ W, __bf16* __restrict__ Wb)
{
    const int idx  = blockIdx.x * 256 + threadIdx.x;  // 0..12287
    const int lane = idx & 31;
    const int kb   = (idx >> 5) & 7;
    const int tile = idx >> 8;
    const int jrow  = tile * 16 + (lane & 15);
    const int kbase = kb * 32 + (lane >> 4) * 16;
    const float* src = W + (size_t)jrow * H_DIM + kbase;
    __bf16* dst = Wb + (size_t)idx * 16;
#pragma unroll
    for (int e = 0; e < 8; ++e) {
        dst[e]     = (__bf16)src[e];        // K = kbase .. kbase+7
        dst[8 + e] = (__bf16)src[8 + e];    // K = kbase+8 .. kbase+15
    }
}

// ---------------------------------------------------------------------------
// Kernel C: sequential GRU scan.
//  - recurrent matvec on the matrix units (24 waves x 2 tiles x 8 WMMAs)
//  - per-step iu row DMA'd into a double-buffered LDS stage by the TDM,
//    overlapped with the previous step's compute; s_wait_tensorcnt + barrier
//    publishes it to the workgroup.
// ---------------------------------------------------------------------------
__global__ void __launch_bounds__(768, 1)
scan_kernel(const float* __restrict__ iu,    // [T, 3H]
            const float* __restrict__ pred,  // [T]
            const __bf16* __restrict__ Wb,   // packed W_hh (WMMA A layout)
            const float* __restrict__ h0,    // [H]
            const float* __restrict__ b_n,   // [H]
            float* __restrict__ out)         // [T]
{
    __shared__ float h_s[H_DIM];
    __shared__ v16bf hbf_v[H_DIM / 16];                // 256 bf16, 32B-aligned
    __shared__ float hu_s[G3];
    __shared__ __align__(16) float iu_lds[2][G3];      // TDM double buffer

    const int j    = threadIdx.x;
    const int lane = j & 31;
    const int wave = j >> 5;                 // 0..23
    const int half = lane >> 4;
    const int nsel = lane & 15;

    // Raw LDS byte addresses for the TDM descriptor (low 32 bits of the
    // generic pointer to LDS are the in-WGP LDS offset).
    const unsigned ldsA[2] = { (unsigned)(size_t)&iu_lds[0][0],
                               (unsigned)(size_t)&iu_lds[1][0] };

    __bf16* hbf = (__bf16*)hbf_v;
    if (j < H_DIM) {
        const float h = h0[j];
        h_s[j] = h;
        hbf[j] = (__bf16)h;
    }
    const float bn_j = (j < H_DIM) ? b_n[j] : 0.0f;

    // Kick off the DMA of the first iu row.
    if (wave == 0) tdm_load_row(ldsA[0], iu);
    __syncthreads();

    const v16bf* wbase = (const v16bf*)Wb;

    for (int t = 0; t < T_STEPS; ++t) {
        // Wave 0 owns TENSORcnt: DMA of row t (issued last step) must be done.
        if (wave == 0) __builtin_amdgcn_s_wait_tensorcnt(0);

        // hu = W_hh @ h : h broadcast into all 16 B-columns, so every output
        // column of the 16x16 D tile equals hu[tile-rows]. K=256 -> 8 WMMAs.
#pragma unroll
        for (int tt = 0; tt < 2; ++tt) {
            const int tile = wave * 2 + tt;
            const v16bf* wrow = wbase + ((size_t)tile * 8) * 32 + lane;
            v8f c = {};
#pragma unroll
            for (int kb = 0; kb < 8; ++kb) {
                const v16bf a = wrow[kb * 32];     // coalesced 32B/lane
                const v16bf b = *(const v16bf*)(hbf + kb * 32 + half * 16);
                c = __builtin_amdgcn_wmma_f32_16x16x32_bf16(
                        false, a, false, b, (short)0, c, false, false);
            }
            if (nsel == 0) {                       // column N=0 lanes scatter hu
                const int jbase = tile * 16 + half * 8;
#pragma unroll
                for (int r = 0; r < 8; ++r) hu_s[jbase + r] = c[r];
            }
        }

        // Overlap: DMA the next step's iu row into the other buffer.
        if (wave == 0 && t + 1 < T_STEPS)
            tdm_load_row(ldsA[(t + 1) & 1], iu + (size_t)(t + 1) * G3);

        __syncthreads();   // hu_s + DMA'd iu row now visible to all waves

        const float* iucur = iu_lds[t & 1];
        if (j < H_DIM) {
            const float r  = sigmoidf_(iucur[j]         + hu_s[j]);
            const float z  = sigmoidf_(iucur[H_DIM + j] + hu_s[H_DIM + j]);
            const float g  = tanhf(iucur[2 * H_DIM + j] +
                                   r * (hu_s[2 * H_DIM + j] + bn_j));
            const float hn = (1.0f - z) * g + z * h_s[j];
            h_s[j] = hn;
            hbf[j] = (__bf16)hn;                   // feed next step's WMMA B
        }
        __syncthreads();

        if (j == 0) out[t] = h_s[0] * pred[t] + h_s[1];
    }
}

// ---------------------------------------------------------------------------
// Launch
// ---------------------------------------------------------------------------
extern "C" void kernel_launch(void* const* d_in, const int* in_sizes, int n_in,
                              void* d_out, int out_size, void* d_ws, size_t ws_size,
                              hipStream_t stream) {
    const float* input_gru    = (const float*)d_in[0]; // [T, IN]
    const float* input_linear = (const float*)d_in[1]; // [T, LIN]
    const float* init_hidden  = (const float*)d_in[2]; // [H]
    const float* W_ih         = (const float*)d_in[3]; // [3H, IN]
    const float* W_hh         = (const float*)d_in[4]; // [3H, H]
    const float* b            = (const float*)d_in[5]; // [3H]
    const float* b_n          = (const float*)d_in[6]; // [H]
    const float* W_lin        = (const float*)d_in[7]; // [1, LIN]
    const float* b_lin        = (const float*)d_in[8]; // [1]
    float* out = (float*)d_out;                        // [T]

    // Workspace layout (all offsets 32B-aligned)
    float*  iu_all = (float*)d_ws;                     // T*3H f32   (192 MB)
    float*  pred   = iu_all + (size_t)T_STEPS * G3;    // T f32
    __bf16* Wb     = (__bf16*)(pred + T_STEPS);        // 48*8*32*16 bf16 (384 KB)

    // One-time pack of W_hh into the WMMA A-operand layout.
    pack_whh_wmma<<<48, 256, 0, stream>>>(W_hh, Wb);

    // Linear branch predictions (independent).
    pred_kernel<<<T_STEPS / 8, 256, 0, stream>>>(input_linear, W_lin, b_lin, pred);

    // Big input-side GEMM on the matrix units.
    dim3 grid(T_STEPS / 16 / 8, G3 / 16);
    gemm_iu_wmma<<<grid, 256, 0, stream>>>(input_gru, W_ih, b, iu_all);

    // Sequential recurrence (same stream -> ordered after the above).
    scan_kernel<<<1, G3, 0, stream>>>(iu_all, pred, Wb, init_hidden, b_n, out);
}
